// GIN_19920058318979
// MI455X (gfx1250) — matrix operand order repaired and checked by reading, not compiled, not run on hardware
//
#include <hip/hip_runtime.h>
#include <hip/hip_bf16.h>
#include <math.h>

#define N_NODES    50000
#define N_EDGES    500000
#define NFEAT      128
#define NCONVS     4
#define NUM_GRAPHS 128
#define NUM_CLASSES 2
#define BN_EPS     1e-5f
#define NE_TOTAL   (N_NODES * NFEAT)   // 6,400,000

typedef __bf16 bf16_t;
typedef __attribute__((ext_vector_type(8)))  bf16_t v8bf;
typedef __attribute__((ext_vector_type(16))) bf16_t v16bf;
typedef __attribute__((ext_vector_type(8)))  float  v8f;

// ---------------- utility ----------------
__global__ void k_zero(float* __restrict__ p, int n) {
    int i = blockIdx.x * blockDim.x + threadIdx.x;
    int stride = gridDim.x * blockDim.x;
    for (; i < n; i += stride) p[i] = 0.0f;
}

// Transpose-convert one 128x128 f32 weight (K-major, [k][n]) to bf16 [n][k]
__global__ void k_wprep(const float* __restrict__ W, bf16_t* __restrict__ Wt) {
    int i = blockIdx.x * blockDim.x + threadIdx.x;   // 16384 threads
    if (i >= NFEAT * NFEAT) return;
    int n = i >> 7, k = i & 127;
    Wt[n * NFEAT + k] = (bf16_t)W[k * NFEAT + n];
}

// ---------------- edge aggregation: agg[dst] += h[src] ----------------
// 32 threads per edge, 4 contiguous floats per thread
__global__ void k_scatter(const float* __restrict__ h, const int* __restrict__ src,
                          const int* __restrict__ dst, float* __restrict__ agg) {
    int gid = blockIdx.x * blockDim.x + threadIdx.x;
    if (gid >= N_EDGES * 32) return;
    int e = gid >> 5;
    int c = (gid & 31) * 4;
    int s = src[e], d = dst[e];
    float4 v = *(const float4*)(h + s * NFEAT + c);
    float* ap = agg + d * NFEAT + c;
    atomicAdd(ap + 0, v.x);
    atomicAdd(ap + 1, v.y);
    atomicAdd(ap + 2, v.z);
    atomicAdd(ap + 3, v.w);
}

// zin = bf16(h + agg)
__global__ void k_addcvt(const float* __restrict__ h, const float* __restrict__ agg,
                         bf16_t* __restrict__ out, int n) {
    int i = blockIdx.x * blockDim.x + threadIdx.x;
    if (i < n) out[i] = (bf16_t)(h[i] + agg[i]);
}

// ---------------- WMMA GEMM: out = relu(A[Mx128] * W[128x128] + bias) ----------------
// A: bf16 row-major MxK (K=128). Bt: bf16 N-major (NxK) pre-transposed weights.
// One block = one 16-row M tile; 8 waves = 8 N tiles of 16. K split into 4 chunks of 32.
template <bool OUT_BF16>
__global__ __launch_bounds__(256) void k_gemm(const bf16_t* __restrict__ A,
                                              const bf16_t* __restrict__ Bt,
                                              const float* __restrict__ bias,
                                              void* __restrict__ outp) {
    const int lane  = threadIdx.x & 31;
    const int hh    = lane >> 4;         // K-half select (A/B VGPR layout)
    const int l15   = lane & 15;
    const int ntile = threadIdx.x >> 5;  // 0..7
    const int mbase = blockIdx.x * 16;
    const int ncol  = ntile * 16 + l15;

    const bf16_t* arow = A + (mbase + l15) * NFEAT;   // row m = lane&15
    const bf16_t* brow = Bt + ncol * NFEAT;           // column n = ncol

    v8f acc = {};
#pragma unroll
    for (int k0 = 0; k0 < 128; k0 += 32) {
        // A 16x32 bf16 fragment: lane half 0 -> K {k0..k0+7, k0+16..k0+23},
        //                        lane half 1 -> K {k0+8..k0+15, k0+24..k0+31}
        const bf16_t* ap = arow + k0 + hh * 8;
        v8bf alo = *(const v8bf*)(ap);
        v8bf ahi = *(const v8bf*)(ap + 16);
        v16bf af = __builtin_shufflevector(alo, ahi, 0,1,2,3,4,5,6,7,8,9,10,11,12,13,14,15);
        // B 32x16 bf16 fragment: lane half 0 -> K {k0..k0+7, k0+16..k0+23},
        //                        lane half 1 -> K {k0+8..k0+15, k0+24..k0+31}
        const bf16_t* bp = brow + k0 + hh * 8;
        v8bf blo = *(const v8bf*)(bp);
        v8bf bhi = *(const v8bf*)(bp + 16);
        v16bf bfrag = __builtin_shufflevector(blo, bhi, 0,1,2,3,4,5,6,7,8,9,10,11,12,13,14,15);

        acc = __builtin_amdgcn_wmma_f32_16x16x32_bf16(
            /*neg_a=*/false, af, /*neg_b=*/false, bfrag,
            /*c_mod=*/(short)0, acc, /*reuse_a=*/false, /*reuse_b=*/false);
    }

    const float bv = bias[ncol];
#pragma unroll
    for (int r = 0; r < 8; ++r) {
        // D layout: lanes 0-15 VGPR r -> row r; lanes 16-31 VGPR r -> row 8+r
        int row = mbase + hh * 8 + r;
        float v = acc[r] + bv;
        v = v > 0.0f ? v : 0.0f;   // ReLU (both GEMMs are followed by ReLU)
        if (OUT_BF16) ((bf16_t*)outp)[row * NFEAT + ncol] = (bf16_t)v;
        else          ((float*)outp)[row * NFEAT + ncol]  = v;
    }
}

// ---------------- BatchNorm ----------------
__global__ void k_bnstats(const float* __restrict__ h, float* __restrict__ sums,
                          float* __restrict__ sqs, int M) {
    int c = threadIdx.x;   // 128 threads = channels (coalesced across c)
    float s = 0.0f, q = 0.0f;
    for (int r = blockIdx.x; r < M; r += gridDim.x) {
        float v = h[r * NFEAT + c];
        s += v;
        q += v * v;
    }
    atomicAdd(&sums[c], s);
    atomicAdd(&sqs[c], q);
}

__global__ void k_bnapply(const float* __restrict__ hin, float* __restrict__ hout,
                          const float* __restrict__ sums, const float* __restrict__ sqs,
                          const float* __restrict__ gamma, const float* __restrict__ beta) {
    int i = blockIdx.x * blockDim.x + threadIdx.x;
    if (i >= NE_TOTAL) return;
    int c = i & 127;
    const float inv = 1.0f / (float)N_NODES;
    float mu  = sums[c] * inv;
    float var = sqs[c] * inv - mu * mu;          // population variance (ddof=0)
    hout[i] = (hin[i] - mu) * rsqrtf(var + BN_EPS) * gamma[c] + beta[c];
}

// ---------------- global mean pool ----------------
__global__ void k_pool(const float* __restrict__ h, const int* __restrict__ batch,
                       float* __restrict__ gsums, float* __restrict__ gcnt) {
    int i = blockIdx.x * blockDim.x + threadIdx.x;
    if (i >= NE_TOTAL) return;
    int node = i >> 7, c = i & 127;
    int g = batch[node];
    atomicAdd(&gsums[g * NFEAT + c], h[i]);
    if (c == 0) atomicAdd(&gcnt[g], 1.0f);
}

// ---------------- head: fc1 -> relu -> fc2 -> log_softmax ----------------
__global__ void k_head(const float* __restrict__ gsums, const float* __restrict__ gcnt,
                       const float* __restrict__ fc1w, const float* __restrict__ fc1b,
                       const float* __restrict__ fc2w, const float* __restrict__ fc2b,
                       float* __restrict__ out) {
    __shared__ float gm[NFEAT];
    __shared__ float p0[NFEAT];
    __shared__ float p1[NFEAT];
    int g = blockIdx.x, j = threadIdx.x;
    float cnt = gcnt[g];
    cnt = cnt > 1.0f ? cnt : 1.0f;
    gm[j] = gsums[g * NFEAT + j] / cnt;
    __syncthreads();
    float acc = fc1b[j];
    for (int k = 0; k < NFEAT; ++k) acc += gm[k] * fc1w[k * NFEAT + j];
    float tv = acc > 0.0f ? acc : 0.0f;
    p0[j] = tv * fc2w[j * NUM_CLASSES + 0];
    p1[j] = tv * fc2w[j * NUM_CLASSES + 1];
    __syncthreads();
    for (int s = 64; s > 0; s >>= 1) {
        if (j < s) { p0[j] += p0[j + s]; p1[j] += p1[j + s]; }
        __syncthreads();
    }
    if (j == 0) {
        float l0 = p0[0] + fc2b[0];
        float l1 = p1[0] + fc2b[1];
        float mx = l0 > l1 ? l0 : l1;
        float lse = mx + logf(expf(l0 - mx) + expf(l1 - mx));
        out[g * 2 + 0] = l0 - lse;
        out[g * 2 + 1] = l1 - lse;
    }
}

// ---------------- launcher ----------------
extern "C" void kernel_launch(void* const* d_in, const int* in_sizes, int n_in,
                              void* d_out, int out_size, void* d_ws, size_t ws_size,
                              hipStream_t stream) {
    const float* x      = (const float*)d_in[0];
    const int*   ei     = (const int*)d_in[1];
    const int*   batch  = (const int*)d_in[2];
    const float* Ws1    = (const float*)d_in[3];
    const float* bs1    = (const float*)d_in[4];
    const float* Ws2    = (const float*)d_in[5];
    const float* bs2    = (const float*)d_in[6];
    const float* gammas = (const float*)d_in[7];
    const float* betas  = (const float*)d_in[8];
    const float* fc1w   = (const float*)d_in[9];
    const float* fc1b   = (const float*)d_in[10];
    const float* fc2w   = (const float*)d_in[11];
    const float* fc2b   = (const float*)d_in[12];
    const int* src = ei;             // edge_index[0]
    const int* dst = ei + N_EDGES;   // edge_index[1]

    // Workspace carve-up (256B aligned). agg aliases hB (pre-BN GEMM2 output).
    char* ws = (char*)d_ws;
    size_t off = 0;
    auto carve = [&](size_t bytes) {
        void* p = ws + off;
        off += (bytes + 255) & ~(size_t)255;
        return p;
    };
    float*  hA    = (float*)carve((size_t)NE_TOTAL * 4);   // persistent node features
    float*  hB    = (float*)carve((size_t)NE_TOTAL * 4);   // agg buffer / pre-BN output
    bf16_t* zin   = (bf16_t*)carve((size_t)NE_TOTAL * 2);  // bf16 GEMM1 input
    bf16_t* t1    = (bf16_t*)carve((size_t)NE_TOTAL * 2);  // bf16 GEMM2 input
    bf16_t* W1t   = (bf16_t*)carve((size_t)NCONVS * NFEAT * NFEAT * 2);
    bf16_t* W2t   = (bf16_t*)carve((size_t)NCONVS * NFEAT * NFEAT * 2);
    float*  stats = (float*)carve(2 * NFEAT * 4);          // [sum | sumsq]
    float*  gsums = (float*)carve((size_t)(NUM_GRAPHS * NFEAT + NUM_GRAPHS) * 4);
    float*  gcnt  = gsums + NUM_GRAPHS * NFEAT;

    // Pre-convert weights (deterministic, re-done every call)
    for (int l = 0; l < NCONVS; ++l) {
        k_wprep<<<64, 256, 0, stream>>>(Ws1 + l * NFEAT * NFEAT, W1t + l * NFEAT * NFEAT);
        k_wprep<<<64, 256, 0, stream>>>(Ws2 + l * NFEAT * NFEAT, W2t + l * NFEAT * NFEAT);
    }

    const int elemBlocks = (NE_TOTAL + 255) / 256;          // 25000
    const int edgeBlocks = (N_EDGES * 32 + 255) / 256;      // 62500
    const int gemmBlocks = N_NODES / 16;                    // 3125 (exact)

    const float* hin = x;
    for (int l = 0; l < NCONVS; ++l) {
        k_zero<<<2048, 256, 0, stream>>>(hB, NE_TOTAL);
        k_scatter<<<edgeBlocks, 256, 0, stream>>>(hin, src, dst, hB);
        k_addcvt<<<elemBlocks, 256, 0, stream>>>(hin, hB, zin, NE_TOTAL);
        k_gemm<true ><<<gemmBlocks, 256, 0, stream>>>(zin, W1t + l * NFEAT * NFEAT,
                                                      bs1 + l * NFEAT, t1);
        k_gemm<false><<<gemmBlocks, 256, 0, stream>>>(t1, W2t + l * NFEAT * NFEAT,
                                                      bs2 + l * NFEAT, hB);
        k_zero<<<1, 256, 0, stream>>>(stats, 2 * NFEAT);
        k_bnstats<<<256, 128, 0, stream>>>(hB, stats, stats + NFEAT, N_NODES);
        k_bnapply<<<elemBlocks, 256, 0, stream>>>(hB, hA, stats, stats + NFEAT,
                                                  gammas + l * NFEAT, betas + l * NFEAT);
        hin = hA;
    }

    k_zero<<<64, 256, 0, stream>>>(gsums, NUM_GRAPHS * NFEAT + NUM_GRAPHS);
    k_pool<<<elemBlocks, 256, 0, stream>>>(hA, batch, gsums, gcnt);
    k_head<<<NUM_GRAPHS, 128, 0, stream>>>(gsums, gcnt, fc1w, fc1b, fc2w, fc2b, (float*)d_out);
}
